// Decoder_21595095564698
// MI455X (gfx1250) — matrix operand order repaired and checked by reading, not compiled, not run on hardware
//
#include <hip/hip_runtime.h>
#include <cmath>

#define BTOT 16384

typedef __attribute__((ext_vector_type(16))) _Float16 v16h;
typedef __attribute__((ext_vector_type(8)))  float    v8f;

union H16 { v16h v; _Float16 e[16]; };
union F8  { v8f  v; float    e[8];  };

__device__ __forceinline__ v8f wmma32(v16h a, v16h b, v8f c) {
  // D = A(16x32,f16) * B(32x16,f16) + C(16x16,f32)
  return __builtin_amdgcn_wmma_f32_16x16x32_f16(false, a, false, b, (short)0, c, false, false);
}

// Fast transcendentals: v_exp_f32 + v_rcp_f32 (TRANS32 ops, co-issue with WMMA),
// avoids the IEEE-div expansion (v_div_scale/v_div_fmas chains).
__device__ __forceinline__ float sigm(float x) {
  return __builtin_amdgcn_rcpf(1.f + __expf(-x));
}
__device__ __forceinline__ float tanh_f(float x) {
  // 1 - 2/(e^{2x}+1); exact limits: x->+inf => 1, x->-inf => -1
  return 1.f - 2.f * __builtin_amdgcn_rcpf(__expf(2.f * x) + 1.f);
}

// Gather an A-fragment (16x32 f16) from an f32 LDS row buffer.
// A layout: lane L holds row M=L&15; group0 (L<16): j0..15 -> K 0..15;
// group1 (L>=16): j0..15 -> K16..31.
__device__ __forceinline__ v16h gatherA(const float* buf, int stride, int lane, int kbase) {
  H16 a;
  const int row = lane & 15;
  const int khalf = (lane & 16);   // 0 or 16
  const float* p = buf + row * stride;
  #pragma unroll
  for (int j = 0; j < 16; ++j) {
    int k = kbase + khalf + j;
    a.e[j] = (_Float16)p[k];
  }
  return a.v;
}

// ---------------- kernel 1: V = mean_l(enc[l,b,:]) @ Wv^T + bv ----------------
// one wave per batch row; 188MB streaming read dominates -> memory bound.
__global__ void __launch_bounds__(256)
vctx_kernel(const float* __restrict__ enc, const float* __restrict__ Wv,
            const float* __restrict__ bv, float* __restrict__ Vout) {
  __shared__ float sm[8][64];
  const int w = threadIdx.x >> 5, lane = threadIdx.x & 31;
  const int b = blockIdx.x * 8 + w;
  float s0 = 0.f, s1 = 0.f;
  for (int l = 0; l < 48; ++l) {
    const float* p = enc + ((size_t)l * BTOT + b) * 60;
    s0 += p[lane];
    if (lane < 28) s1 += p[lane + 32];
  }
  sm[w][lane] = s0 * (1.f / 48.f);
  if (lane < 28) sm[w][lane + 32] = s1 * (1.f / 48.f);
  __syncthreads();
  if (lane < 30) {
    float acc = bv[lane];
    const float* wr = Wv + lane * 60;
    for (int e = 0; e < 60; ++e) acc += sm[w][e] * wr[e];
    Vout[(size_t)b * 30 + lane] = acc;
  }
}

// ---------------- kernel 2: 12-step recurrence, one wave = 16 batch rows -------
__global__ void __launch_bounds__(128)
decoder_kernel(const float* __restrict__ y, const float* __restrict__ hidden,
               const float* __restrict__ Vg,
               const float* __restrict__ Wg,  const float* __restrict__ bg,
               const float* __restrict__ Wih, const float* __restrict__ bih,
               const float* __restrict__ Whh, const float* __restrict__ bhh,
               const float* __restrict__ Wf1, const float* __restrict__ bf1,
               const float* __restrict__ Wf2, const float* __restrict__ bf2,
               const float* __restrict__ Wf3, const float* __restrict__ bf3,
               float* __restrict__ dout) {
  // per-wave LDS state (odd dword strides -> conflict-free row access)
  __shared__ float s_g[4][16 * 65];   // [h(0..29)|0,0|V(32..61)|0,0]
  __shared__ float s_x[4][16 * 65];   // [rin(0..3)|enh(4..33)|h(34..63)]
  __shared__ float s_z1[4][16 * 33];  // zfc1 (cols 30,31 stay 0)
  __shared__ float s_z2[4][16 * 33];  // zfc2 (cols 15..31 stay 0)

  const int w    = threadIdx.x >> 5;
  const int lane = threadIdx.x & 31;
  const int colb = lane & 15;
  const int hi   = (lane >> 4) & 1;
  const int rowbase = hi * 8;
  const int b0 = (blockIdx.x * 4 + w) * 16;

  float* gbuf = s_g[w];
  float* xbuf = s_x[w];
  float* zb1  = s_z1[w];
  float* zb2  = s_z2[w];

  for (int idx = lane; idx < 16 * 65; idx += 32) { gbuf[idx] = 0.f; xbuf[idx] = 0.f; }
  for (int idx = lane; idx < 16 * 33; idx += 32) { zb1[idx] = 0.f;  zb2[idx] = 0.f; }

  // initial state into LDS (lanes 0..15: one row each)
  if (lane < 16) {
    const int b = b0 + lane;
    const float* hp = hidden + (size_t)b * 30;
    const float* vp = Vg + (size_t)b * 30;
    float* gr = gbuf + lane * 65;
    float* xr = xbuf + lane * 65;
    #pragma unroll
    for (int d = 0; d < 30; ++d) {
      float hv = hp[d];
      gr[d] = hv; xr[34 + d] = hv; gr[32 + d] = vp[d];
    }
    const float* yp = y + (size_t)b * 52;   // y[b][0][:]
    #pragma unroll
    for (int d = 0; d < 4; ++d) { float v = yp[d]; xr[d] = (v != v) ? 0.f : v; }
  }

  // ---- B fragments (register resident; B layout: K=(lane&16)+j, N=lane&15) ----
  H16 Bg_[2][2];
  #pragma unroll
  for (int ks = 0; ks < 2; ++ks)
    #pragma unroll
    for (int nt = 0; nt < 2; ++nt) {
      const int n = nt * 16 + colb;
      #pragma unroll
      for (int j = 0; j < 16; ++j) {
        const int k = ks * 32 + (lane & 16) + j;
        float wv = 0.f;
        if (n < 30) {
          if (k < 30) wv = Wg[n * 60 + k];                     // h part
          else if (k >= 32 && k < 62) wv = Wg[n * 60 + k - 2]; // V part
        }
        Bg_[ks][nt].e[j] = (_Float16)wv;
      }
    }

  H16 Bl_[2][8];   // fused [W_ih|W_hh], gates padded 30->32 cols
  #pragma unroll
  for (int t = 0; t < 8; ++t) {
    const int nc = t * 16 + colb;
    const int q = nc >> 5, m = nc & 31;
    #pragma unroll
    for (int ks = 0; ks < 2; ++ks)
      #pragma unroll
      for (int j = 0; j < 16; ++j) {
        const int k = ks * 32 + (lane & 16) + j;
        float wv = 0.f;
        if (m < 30) {
          const int orow = q * 30 + m;
          wv = (k < 34) ? Wih[orow * 34 + k] : Whh[orow * 30 + (k - 34)];
        }
        Bl_[ks][t].e[j] = (_Float16)wv;
      }
  }

  H16 Bf1_[2];
  #pragma unroll
  for (int nt = 0; nt < 2; ++nt) {
    const int n = nt * 16 + colb;
    #pragma unroll
    for (int j = 0; j < 16; ++j) {
      const int k = (lane & 16) + j;
      Bf1_[nt].e[j] = (_Float16)((n < 30 && k < 30) ? Wf1[n * 31 + k] : 0.f);
    }
  }
  H16 Bf2_;
  #pragma unroll
  for (int j = 0; j < 16; ++j) {
    const int k = (lane & 16) + j;
    Bf2_.e[j] = (_Float16)((colb < 15 && k < 30) ? Wf2[colb * 30 + k] : 0.f);
  }
  H16 Bf3_;
  #pragma unroll
  for (int j = 0; j < 16; ++j) {
    const int k = (lane & 16) + j;
    Bf3_.e[j] = (_Float16)((colb == 0 && k < 15) ? Wf3[k] : 0.f);
  }

  // ---- per-lane biases / C-layout state fragments ----
  float bg_s[2], bf1_s[2], wf1l_s[2];
  #pragma unroll
  for (int sub = 0; sub < 2; ++sub) {
    const int c = sub * 16 + colb;
    bg_s[sub]   = (c < 30) ? bg[c] : 0.f;
    bf1_s[sub]  = (c < 30) ? bf1[c] : 0.f;
    wf1l_s[sub] = (c < 30) ? Wf1[c * 31 + 30] : 0.f;  // se column folded into bias
  }
  float bL_s[8];
  #pragma unroll
  for (int t = 0; t < 8; ++t) {
    const int nc = t * 16 + colb;
    const int q = nc >> 5, m = nc & 31;
    bL_s[t] = (m < 30) ? (bih[q * 30 + m] + bhh[q * 30 + m]) : 0.f;
  }
  const float bf2_s = (colb < 15) ? bf2[colb] : 0.f;
  const float bf3_s = bf3[0];

  F8 hf[2], cf[2], vf[2];
  #pragma unroll
  for (int sub = 0; sub < 2; ++sub) {
    const int c = sub * 16 + colb;
    #pragma unroll
    for (int r = 0; r < 8; ++r) {
      const int b = b0 + rowbase + r;
      hf[sub].e[r] = (c < 30) ? hidden[(size_t)b * 30 + c] : 0.f;
      vf[sub].e[r] = (c < 30) ? Vg[(size_t)b * 30 + c] : 0.f;
      cf[sub].e[r] = 0.f;
    }
  }

  // ---------------- 12-step recurrence ----------------
  for (int i = 1; i <= 12; ++i) {
    const float se = (float)i * (1.f / 12.f);

    // gate = sigmoid([h|V] @ Wg^T + bg)
    v16h aG0 = gatherA(gbuf, 65, lane, 0);
    v16h aG1 = gatherA(gbuf, 65, lane, 32);
    F8 gt[2];
    #pragma unroll
    for (int nt = 0; nt < 2; ++nt) {
      v8f acc = {};
      acc = wmma32(aG0, Bg_[0][nt].v, acc);
      acc = wmma32(aG1, Bg_[1][nt].v, acc);
      gt[nt].v = acc;
    }
    // enh = g*h + (1-g)*V  -> xbuf cols 4..33
    #pragma unroll
    for (int sub = 0; sub < 2; ++sub) {
      const int c = sub * 16 + colb;
      #pragma unroll
      for (int r = 0; r < 8; ++r) {
        const float g = sigm(gt[sub].e[r] + bg_s[sub]);
        const float enh = g * hf[sub].e[r] + (1.f - g) * vf[sub].e[r];
        if (c < 30) xbuf[(rowbase + r) * 65 + 4 + c] = enh;
      }
    }

    // fused LSTM GEMM: [rin|enh|h](K=64) @ [W_ih|W_hh]^T (N=128, gate-padded)
    v16h aL0 = gatherA(xbuf, 65, lane, 0);
    v16h aL1 = gatherA(xbuf, 65, lane, 32);
    F8 acc[8];
    #pragma unroll
    for (int t = 0; t < 8; ++t) {
      v8f a = {};
      a = wmma32(aL0, Bl_[0][t].v, a);
      a = wmma32(aL1, Bl_[1][t].v, a);
      acc[t].v = a;
    }
    // LSTM nonlinearity: gates i,f,g,o at tiles {sub, 2+sub, 4+sub, 6+sub}
    #pragma unroll
    for (int sub = 0; sub < 2; ++sub) {
      const int c = sub * 16 + colb;
      #pragma unroll
      for (int r = 0; r < 8; ++r) {
        const float ig = acc[0 + sub].e[r] + bL_s[0 + sub];
        const float fg = acc[2 + sub].e[r] + bL_s[2 + sub];
        const float gg = acc[4 + sub].e[r] + bL_s[4 + sub];
        const float og = acc[6 + sub].e[r] + bL_s[6 + sub];
        const float cn = sigm(fg) * cf[sub].e[r] + sigm(ig) * tanh_f(gg);
        const float hn = sigm(og) * tanh_f(cn);
        cf[sub].e[r] = cn;
        hf[sub].e[r] = hn;
        if (c < 30) {
          gbuf[(rowbase + r) * 65 + c] = hn;       // gate input next step
          xbuf[(rowbase + r) * 65 + 34 + c] = hn;  // LSTM h input next step
        }
      }
    }

    // zfc1 = relu(h_new @ Wf1[:,:30]^T + bf1 + se*Wf1[:,30])
    v16h aF1 = gatherA(gbuf, 65, lane, 0);
    #pragma unroll
    for (int nt = 0; nt < 2; ++nt) {
      v8f a = {};
      a = wmma32(aF1, Bf1_[nt].v, a);
      F8 z; z.v = a;
      const int c = nt * 16 + colb;
      #pragma unroll
      for (int r = 0; r < 8; ++r) {
        const float zz = fmaxf(z.e[r] + bf1_s[nt] + se * wf1l_s[nt], 0.f);
        if (c < 30) zb1[(rowbase + r) * 33 + c] = zz;
      }
    }

    // zfc2 = relu(zfc1 @ Wf2^T + bf2)
    v16h aF2 = gatherA(zb1, 33, lane, 0);
    {
      v8f a = {};
      a = wmma32(aF2, Bf2_.v, a);
      F8 z; z.v = a;
      #pragma unroll
      for (int r = 0; r < 8; ++r) {
        const float zz = fmaxf(z.e[r] + bf2_s, 0.f);
        if (colb < 15) zb2[(rowbase + r) * 33 + colb] = zz;
      }
    }

    // out = zfc2 @ Wf3^T + bf3  (N=1 WMMA; result in lanes 0/16, col 0)
    v16h aF3 = gatherA(zb2, 33, lane, 0);
    {
      v8f a = {};
      a = wmma32(aF3, Bf3_.v, a);
      F8 o; o.v = a;
      if (colb == 0) {
        #pragma unroll
        for (int r = 0; r < 8; ++r) {
          const float ov = o.e[r] + bf3_s;
          const int b = b0 + rowbase + r;
          dout[(size_t)(i - 1) * BTOT + b] = ov;
          xbuf[(rowbase + r) * 65 + 3] = ov;     // rin_next[3]
        }
      }
    }

    // rin_next[0..2] = y[b][i][1..3]
    if (lane < 16) {
      const float* yp = y + (size_t)(b0 + lane) * 52 + i * 4;
      float* xr = xbuf + lane * 65;
      xr[0] = yp[1]; xr[1] = yp[2]; xr[2] = yp[3];
    }
  }
}

extern "C" void kernel_launch(void* const* d_in, const int* in_sizes, int n_in,
                              void* d_out, int out_size, void* d_ws, size_t ws_size,
                              hipStream_t stream) {
  const float* y      = (const float*)d_in[1];
  const float* enc    = (const float*)d_in[2];
  const float* hidden = (const float*)d_in[3];
  // d_in[4..7] (Wq,bq,Wk,bk) and d_in[10..13] (Wsa*) are dead: softmax over size-1 axis -> attn==1
  const float* Wv  = (const float*)d_in[8];
  const float* bv  = (const float*)d_in[9];
  const float* Wg  = (const float*)d_in[14];
  const float* bg  = (const float*)d_in[15];
  const float* Wf1 = (const float*)d_in[16];
  const float* bf1 = (const float*)d_in[17];
  const float* Wf2 = (const float*)d_in[18];
  const float* bf2 = (const float*)d_in[19];
  const float* Wf3 = (const float*)d_in[20];
  const float* bf3 = (const float*)d_in[21];
  const float* Wih = (const float*)d_in[22];
  const float* bih = (const float*)d_in[23];
  const float* Whh = (const float*)d_in[24];
  const float* bhh = (const float*)d_in[25];

  float* Vbuf = (float*)d_ws;  // 16384*30 floats

  vctx_kernel<<<BTOT / 8, 256, 0, stream>>>(enc, Wv, bv, Vbuf);
  decoder_kernel<<<(BTOT / 16) / 4, 128, 0, stream>>>(
      y, hidden, Vbuf, Wg, bg, Wih, bih, Whh, bhh,
      Wf1, bf1, Wf2, bf2, Wf3, bf3, (float*)d_out);
}
